// MultiHeadAttention_84679575208070
// MI455X (gfx1250) — matrix operand order repaired and checked by reading, not compiled, not run on hardware
//
#include <hip/hip_runtime.h>
#include <hip/hip_bf16.h>

// ---------------------------------------------------------------------------
// MI455X (gfx1250) multi-head attention, bf16 WMMA pipeline, fp32 I/O.
// ---------------------------------------------------------------------------

typedef __attribute__((ext_vector_type(16))) __bf16 v16bf;
typedef __attribute__((ext_vector_type(8)))  __bf16 v8bf;
typedef __attribute__((ext_vector_type(8)))  float  v8f;

#define B_SZ   2
#define S_LEN  2048
#define HID    1024
#define NH     16
#define HD     64
#define MTOT   (B_SZ * S_LEN)          // 4096 rows

__device__ __forceinline__ v8f wmma_bf16(v16bf a, v16bf b, v8f c) {
  // v_wmma_f32_16x16x32_bf16  D = A(16x32) * B(32x16) + C
  return __builtin_amdgcn_wmma_f32_16x16x32_bf16(
      /*neg_a=*/false, a, /*neg_b=*/false, b,
      /*c_mod=*/(short)0, c, /*reuse_a=*/false, /*reuse_b=*/false);
}

// A-fragment: 16(M)x32(K), row-major source with stride lda.
// Lane l: M = l&15, half = l>>4. elems 0..7 = K[half*8 .. +7],
// elems 8..15 = K[16+half*8 .. +7]   (ISA 7.12.2, 16-bit A 16x32)
__device__ __forceinline__ v16bf load_a_frag(const __bf16* __restrict__ base,
                                             int lda, int m0, int k0, int lane) {
  int row  = m0 + (lane & 15);
  int half = lane >> 4;
  const __bf16* p = base + row * lda + k0 + half * 8;
  v8bf lo = *(const v8bf*)p;
  v8bf hi = *(const v8bf*)(p + 16);
  v16bf a;
#pragma unroll
  for (int i = 0; i < 8; ++i) { a[i] = lo[i]; a[8 + i] = hi[i]; }
  return a;
}

// B-fragment: 32(K)x16(N) where the source is B^T stored row-major
// (N rows x K cols, stride ldb). Lane l: N = l&15, half = l>>4,
// elems 0..7 = K[half*8 .. +7], elems 8..15 = K[16+half*8 .. +7].
__device__ __forceinline__ v16bf load_b_frag(const __bf16* __restrict__ baseT,
                                             int ldb, int n0, int k0, int lane) {
  int col  = n0 + (lane & 15);
  int half = lane >> 4;
  const __bf16* p = baseT + col * ldb + k0 + half * 8;
  v8bf lo = *(const v8bf*)p;
  v8bf hi = *(const v8bf*)(p + 16);
  v16bf b;
#pragma unroll
  for (int i = 0; i < 8; ++i) { b[i] = lo[i]; b[8 + i] = hi[i]; }
  return b;
}

// ---------------------------------------------------------------------------
// fp32 -> bf16 convert
// ---------------------------------------------------------------------------
__global__ void cvt_f32_bf16(const float* __restrict__ src,
                             __bf16* __restrict__ dst, int n) {
  int i = blockIdx.x * blockDim.x + threadIdx.x;
  if (i < n) dst[i] = (__bf16)src[i];
}

// ---------------------------------------------------------------------------
// GEMM: C[M,N] = A[M,K] * W^T, W given row-major as [N,K] (== B^T layout).
// One wave -> 32x64 output (2 A-fragments share 4 B-fragments -> 8 WMMAs
// per K-step); 4 waves/block cover a 128x64 tile.
// ---------------------------------------------------------------------------
template <typename OutT>
__global__ __launch_bounds__(128)
void gemm_wmma(const __bf16* __restrict__ A, const __bf16* __restrict__ Wt,
               OutT* __restrict__ C, int M, int N, int K) {
  int lane = threadIdx.x & 31;
  int wid  = threadIdx.x >> 5;
  int m0   = blockIdx.x * 128 + wid * 32;
  int nb   = blockIdx.y * 64;

  v8f acc[8];
#pragma unroll
  for (int j = 0; j < 8; ++j)
#pragma unroll
    for (int r = 0; r < 8; ++r) acc[j][r] = 0.0f;

  for (int k0 = 0; k0 < K; k0 += 32) {
    v16bf a0 = load_a_frag(A, K, m0,      k0, lane);
    v16bf a1 = load_a_frag(A, K, m0 + 16, k0, lane);
    // speculative prefetch of the next K slab (global_prefetch_b8)
    __builtin_prefetch(A + (m0 + (lane & 15)) * K + k0 + 128, 0, 1);
#pragma unroll
    for (int j = 0; j < 4; ++j) {
      v16bf b = load_b_frag(Wt, K, nb + j * 16, k0, lane);
      __builtin_prefetch(Wt + (nb + j * 16 + (lane & 15)) * K + k0 + 128, 0, 1);
      acc[j]     = wmma_bf16(a0, b, acc[j]);
      acc[4 + j] = wmma_bf16(a1, b, acc[4 + j]);
    }
  }

  // C/D layout: lane l: N = l&15, rows M = r + 8*(l>>4), r = vgpr idx
  int ln = lane & 15, half = lane >> 4;
#pragma unroll
  for (int j = 0; j < 4; ++j)
#pragma unroll
    for (int r = 0; r < 8; ++r) {
      C[(m0 + r + 8 * half) * N + nb + j * 16 + ln]      = (OutT)acc[j][r];
      C[(m0 + 16 + r + 8 * half) * N + nb + j * 16 + ln] = (OutT)acc[4 + j][r];
    }
}

// ---------------------------------------------------------------------------
// RoPE + relayout: q/k/v flat [b,s,o] bf16 ->
//   Qr,Kr: [b,h,s,d] bf16 (rotated), Vt: [b,h,d,s] bf16 (transposed)
// ---------------------------------------------------------------------------
__global__ void rope_relayout(const __bf16* __restrict__ qt,
                              const __bf16* __restrict__ kt,
                              const __bf16* __restrict__ vt,
                              __bf16* __restrict__ Qr,
                              __bf16* __restrict__ Kr,
                              __bf16* __restrict__ Vt) {
  int idx = blockIdx.x * blockDim.x + threadIdx.x;   // over B*S*HID
  if (idx >= B_SZ * S_LEN * HID) return;
  int o = idx & (HID - 1);
  int s = (idx >> 10) & (S_LEN - 1);
  int b = idx >> 21;
  int h = o >> 6, d = o & 63;
  int i = d & 31;
  // inv_freq = 10000^(-i/32) = exp(-i * ln(1e4)/32)
  float ang = (float)s * __expf(-(float)i * (9.210340371976184f / 32.0f));
  float c = __cosf(ang), sn = __sinf(ang);
  int op = (d < 32) ? (o + 32) : (o - 32);
  int pidx = idx - o + op;
  float qv = (float)qt[idx], qp = (float)qt[pidx];
  float kv = (float)kt[idx], kp = (float)kt[pidx];
  float rotq = (d < 32) ? -qp : qp;
  float rotk = (d < 32) ? -kp : kp;
  int dst = ((b * NH + h) * S_LEN + s) * HD + d;
  Qr[dst] = (__bf16)(qv * c + rotq * sn);
  Kr[dst] = (__bf16)(kv * c + rotk * sn);
  Vt[((b * NH + h) * HD + d) * S_LEN + s] = vt[idx];
}

// ---------------------------------------------------------------------------
// Flash attention: one wave = 16 query rows, streams 32 keys / iteration.
// QK^T: 4 WMMAs (2 tiles x K=64); PV: 4 WMMAs (4 d-tiles, K=32 keys).
// Softmax row ops go through a wave-private LDS transpose (wide DS ops)
// instead of lane butterflies: each lane owns half a row (16 cols) in-register.
// ---------------------------------------------------------------------------
__global__ __launch_bounds__(128)
void attn_kernel(const __bf16* __restrict__ Qr, const __bf16* __restrict__ Kr,
                 const __bf16* __restrict__ Vt, const float* __restrict__ amask,
                 __bf16* __restrict__ ctx) {
  __shared__ float  ldsS[4][16 * 32];   // raw S tile, fp32 (2KB/wave)
  __shared__ __bf16 ldsP[4][16 * 32];   // P tile, bf16   (1KB/wave)
  __shared__ float  ldsA[4][16];        // per-row alpha / 1/l broadcast

  int lane = threadIdx.x & 31;
  int wid  = threadIdx.x >> 5;
  int b = blockIdx.z, h = blockIdx.y;
  int q0 = blockIdx.x * 64 + wid * 16;
  int ln = lane & 15, half = lane >> 4;

  const __bf16* Qb = Qr + ((b * NH + h) * S_LEN) * HD;
  const __bf16* Kb = Kr + ((b * NH + h) * S_LEN) * HD;
  const __bf16* Vb = Vt + ((b * NH + h) * HD) * S_LEN;
  float*  myS = &ldsS[wid][0];
  __bf16* myP = &ldsP[wid][0];
  float*  myA = &ldsA[wid][0];

  // Q A-fragments, K-split d=0..31 / 32..63
  v16bf aq0 = load_a_frag(Qb, HD, q0, 0, lane);
  v16bf aq1 = load_a_frag(Qb, HD, q0, 32, lane);

  v8f acc[4];
#pragma unroll
  for (int j = 0; j < 4; ++j)
#pragma unroll
    for (int r = 0; r < 8; ++r) acc[j][r] = 0.0f;

  float m_run = -__builtin_inff();      // this lane's row = ln
  float l_run = 0.0f;
  const float scale = 0.125f;           // 64^-0.5

  for (int key0 = 0; key0 < S_LEN; key0 += 32) {
    // ---- S = scale * Q K^T + bias, C layout (bias per-lane column) ----
#pragma unroll
    for (int t = 0; t < 2; ++t) {
      int kt0 = key0 + t * 16;
      v16bf bk0 = load_b_frag(Kb, HD, kt0, 0, lane);
      v16bf bk1 = load_b_frag(Kb, HD, kt0, 32, lane);
      v8f st;
#pragma unroll
      for (int r = 0; r < 8; ++r) st[r] = 0.0f;
      st = wmma_bf16(aq0, bk0, st);
      st = wmma_bf16(aq1, bk1, st);
      float bias = (1.0f - amask[b * S_LEN + kt0 + ln]) * -3.0e38f;
      // stage to LDS row-major: element (row r+8*half, col t*16+ln)
#pragma unroll
      for (int r = 0; r < 8; ++r)
        myS[(r + 8 * half) * 32 + t * 16 + ln] = st[r] * scale + bias;
    }
    asm volatile("s_wait_dscnt 0" ::: "memory");

    // ---- read back transposed: lane owns row ln, cols half*16..+15 ----
    const v8f* rowp = (const v8f*)(myS + ln * 32 + half * 16);
    v8f s0 = rowp[0];
    v8f s1 = rowp[1];

    float mcur = s0[0];
#pragma unroll
    for (int i = 1; i < 8; ++i) mcur = fmaxf(mcur, s0[i]);
#pragma unroll
    for (int i = 0; i < 8; ++i) mcur = fmaxf(mcur, s1[i]);
    mcur = fmaxf(mcur, __shfl_xor(mcur, 16, 32));   // combine halves

    float mnew  = fmaxf(m_run, mcur);
    float alpha = __expf(m_run - mnew);
    m_run = mnew;

    float lsum = 0.0f;
#pragma unroll
    for (int i = 0; i < 8; ++i) { s0[i] = __expf(s0[i] - mnew); lsum += s0[i]; }
#pragma unroll
    for (int i = 0; i < 8; ++i) { s1[i] = __expf(s1[i] - mnew); lsum += s1[i]; }
    lsum += __shfl_xor(lsum, 16, 32);
    l_run = l_run * alpha + lsum;

    // ---- broadcast alpha by row through LDS; store P (A-layout rows) ----
    if (half == 0) myA[ln] = alpha;
    v8bf pb0, pb1;
#pragma unroll
    for (int i = 0; i < 8; ++i) { pb0[i] = (__bf16)s0[i]; pb1[i] = (__bf16)s1[i]; }
    v8bf* pw = (v8bf*)(myP + ln * 32 + half * 16);
    pw[0] = pb0;
    pw[1] = pb1;
    asm volatile("s_wait_dscnt 0" ::: "memory");

    // per-row alpha for accumulator rescale: rows r+8*half
    v8f al = *(const v8f*)(myA + 8 * half);
#pragma unroll
    for (int j = 0; j < 4; ++j)
#pragma unroll
      for (int r = 0; r < 8; ++r) acc[j][r] *= al[r];

    // reload P as an A-fragment (16x32, K = key index)
    const __bf16* pp = myP + ln * 32 + half * 8;
    v8bf plo = *(const v8bf*)pp;
    v8bf phi = *(const v8bf*)(pp + 16);
    v16bf pa;
#pragma unroll
    for (int i = 0; i < 8; ++i) { pa[i] = plo[i]; pa[8 + i] = phi[i]; }

    // ---- PV: B^T rows are Vt[d][keys] -> contiguous 32B loads ----
#pragma unroll
    for (int j = 0; j < 4; ++j) {
      v16bf bv = load_b_frag(Vb, S_LEN, j * 16, key0, lane);
      acc[j] = wmma_bf16(pa, bv, acc[j]);
    }
  }

  // ---- epilogue: broadcast 1/l by row, normalize, write ctx ----
  if (half == 0) myA[ln] = 1.0f / l_run;
  asm volatile("s_wait_dscnt 0" ::: "memory");
  v8f rl = *(const v8f*)(myA + 8 * half);
#pragma unroll
  for (int j = 0; j < 4; ++j)
#pragma unroll
    for (int r = 0; r < 8; ++r) {
      int row = q0 + r + 8 * half;
      int o = h * HD + j * 16 + ln;
      ctx[(b * S_LEN + row) * HID + o] = (__bf16)(acc[j][r] * rl[r]);
    }
}

// ---------------------------------------------------------------------------
// launch
// ---------------------------------------------------------------------------
extern "C" void kernel_launch(void* const* d_in, const int* in_sizes, int n_in,
                              void* d_out, int out_size, void* d_ws, size_t ws_size,
                              hipStream_t stream) {
  const float* x  = (const float*)d_in[0];
  const float* am = (const float*)d_in[1];
  const float* Wq = (const float*)d_in[2];
  const float* Wk = (const float*)d_in[3];
  const float* Wv = (const float*)d_in[4];
  const float* Wo = (const float*)d_in[5];
  float* out = (float*)d_out;

  const int NX = MTOT * HID;          // 4,194,304
  const int NW = HID * HID;           // 1,048,576

  __bf16* ws  = (__bf16*)d_ws;
  __bf16* xb  = ws;                   // NX
  __bf16* wqb = xb  + NX;             // NW
  __bf16* wkb = wqb + NW;
  __bf16* wvb = wkb + NW;
  __bf16* wob = wvb + NW;
  __bf16* qt  = wob + NW;             // NX each
  __bf16* kt  = qt  + NX;
  __bf16* vt  = kt  + NX;
  __bf16* Qr  = vt  + NX;
  __bf16* Kr  = Qr  + NX;
  __bf16* Vt  = Kr  + NX;
  __bf16* ctx = Vt  + NX;             // total 8*NX + 4*NW bf16 = ~75 MB

  // 1) converts
  cvt_f32_bf16<<<(NX + 255) / 256, 256, 0, stream>>>(x, xb, NX);
  cvt_f32_bf16<<<(NW + 255) / 256, 256, 0, stream>>>(Wq, wqb, NW);
  cvt_f32_bf16<<<(NW + 255) / 256, 256, 0, stream>>>(Wk, wkb, NW);
  cvt_f32_bf16<<<(NW + 255) / 256, 256, 0, stream>>>(Wv, wvb, NW);
  cvt_f32_bf16<<<(NW + 255) / 256, 256, 0, stream>>>(Wo, wob, NW);

  // 2) QKV projections (X * W^T)
  dim3 gg(MTOT / 128, HID / 64);
  gemm_wmma<__bf16><<<gg, 128, 0, stream>>>(xb, wqb, qt, MTOT, HID, HID);
  gemm_wmma<__bf16><<<gg, 128, 0, stream>>>(xb, wkb, kt, MTOT, HID, HID);
  gemm_wmma<__bf16><<<gg, 128, 0, stream>>>(xb, wvb, vt, MTOT, HID, HID);

  // 3) RoPE + relayout
  rope_relayout<<<(NX + 255) / 256, 256, 0, stream>>>(qt, kt, vt, Qr, Kr, Vt);

  // 4) attention
  dim3 ga(S_LEN / 64, NH, B_SZ);
  attn_kernel<<<ga, 128, 0, stream>>>(Qr, Kr, Vt, am, ctx);

  // 5) output projection -> fp32
  gemm_wmma<float><<<gg, 128, 0, stream>>>(ctx, wob, out, MTOT, HID, HID);
}
